// DisentangledSelfAttention_60249801228962
// MI455X (gfx1250) — matrix operand order repaired and checked by reading, not compiled, not run on hardware
//
#include <hip/hip_runtime.h>

typedef __bf16 bf16;
typedef __attribute__((ext_vector_type(16))) __bf16 v16bf;
typedef __attribute__((ext_vector_type(8)))  __bf16 v8bf;
typedef __attribute__((ext_vector_type(8)))  float  v8f;
typedef __attribute__((ext_vector_type(4)))  int    v4i;

#define B_   1024
#define S_   100
#define E_   128
#define A_   256
#define H_   4
#define HD_  64
#define SP   112   // S padded to 16-multiple
#define KP   128   // key-dim padded for GEMM2 (4 chunks of 32)
#define XS   136   // proj LDS row stride (halfs): 272B, 16B-aligned, bank-skewed

// ---- gfx1250 async global->LDS DMA (ASYNCcnt-tracked), guarded by __has_builtin ----
#if defined(__has_builtin)
#if __has_builtin(__builtin_amdgcn_global_load_async_to_lds_b128)
#define HAVE_ASYNC_LDS 1
#endif
#endif
#ifndef HAVE_ASYNC_LDS
#define HAVE_ASYNC_LDS 0
#endif

#if HAVE_ASYNC_LDS
typedef __attribute__((address_space(1))) v4i gv4i;   // global (prints as __device__*)
typedef __attribute__((address_space(3))) v4i lv4i;   // LDS    (prints as __shared__*)
__device__ __forceinline__ void async_copy16(const void* g, void* l) {
  __builtin_amdgcn_global_load_async_to_lds_b128((gv4i*)g, (lv4i*)l, 0, 0);
}
__device__ __forceinline__ void wait_async0() {
#if __has_builtin(__builtin_amdgcn_s_wait_asynccnt)
  __builtin_amdgcn_s_wait_asynccnt(0);
#else
  asm volatile("s_wait_asynccnt 0" ::: "memory");
#endif
}
#endif

// ---------------- WMMA helpers ----------------
__device__ __forceinline__ v8f wmma_bf16(v16bf a, v16bf b, v8f c) {
  return __builtin_amdgcn_wmma_f32_16x16x32_bf16(
      false, a, false, b, (short)0, c, false, false);
}

__device__ __forceinline__ v16bf frag2(const bf16* p0, const bf16* p1) {
  v8bf x = *(const v8bf*)p0;   // 16B aligned
  v8bf y = *(const v8bf*)p1;
  v16bf r;
#pragma unroll
  for (int i = 0; i < 8; ++i) { r[i] = x[i]; r[i + 8] = y[i]; }
  return r;
}

// A fragment (16x32 bf16) from a row-major tile; `row` points at row (lane&15).
__device__ __forceinline__ v16bf fragA(const bf16* row, int kc, int lane) {
  int base = kc * 32 + ((lane < 16) ? 0 : 8);
  return frag2(row + base, row + base + 16);
}

// B fragment (32x16 bf16) from an n-major tile (contraction dim contiguous).
__device__ __forceinline__ v16bf fragB(const bf16* mat, int stride, int nt, int kc, int lane) {
  const bf16* p = mat + (size_t)(nt * 16 + (lane & 15)) * stride
                + kc * 32 + ((lane < 16) ? 0 : 8);
  return frag2(p, p + 16);
}

// ---------------- Kernel 1: projections (templated: no runtime branch) ----------------
// G: 0=query@Wq->Qh, 1=key@Wk->Kh, 2=value@Wv->Vh, 3=query@Wres+bres->d_out
template <int G>
__global__ __launch_bounds__(256)
void proj_kernel(const float* __restrict__ X, const float* __restrict__ W,
                 const float* __restrict__ bres,
                 bf16* __restrict__ dstH, float* __restrict__ outRes) {
  extern __shared__ char smem[];
  bf16* Xl = (bf16*)smem;                     // [128][XS]
  bf16* Wt = (bf16*)(smem + 128 * XS * 2);    // [256][XS] : W transposed (n-major)

  const int tid  = threadIdx.x;
  const int wave = tid >> 5, lane = tid & 31;
  const int m0   = blockIdx.x * 128;

  for (int idx = tid; idx < 128 * 128; idx += 256) {
    int r = idx >> 7, e = idx & 127;
    Xl[r * XS + e] = (bf16)X[(size_t)(m0 + r) * E_ + e];
  }
  for (int idx = tid; idx < 128 * 256; idx += 256) {
    int e = idx >> 8, n = idx & 255;
    Wt[n * XS + e] = (bf16)W[(size_t)e * A_ + n];     // transpose into LDS
  }
  __syncthreads();

  // hoist per-r output row offsets out of the nt loop (kills the per-elem div)
  size_t rowoff[8];
#pragma unroll
  for (int r = 0; r < 8; ++r) {
    int m = m0 + wave * 16 + r + ((lane < 16) ? 0 : 8);
    if (G < 3) {
      int bb = m / S_, ss = m % S_;
      rowoff[r] = ((size_t)bb * S_ + ss) * HD_;       // + head*B*S*HD + d later
    } else {
      rowoff[r] = (size_t)m * A_;
    }
  }

  const bf16* arow = Xl + (size_t)(wave * 16 + (lane & 15)) * XS;
  for (int nt = 0; nt < 16; ++nt) {
    v8f acc = {};
#pragma unroll
    for (int kc = 0; kc < 4; ++kc) {
      v16bf a = fragA(arow, kc, lane);
      v16bf b = fragB(Wt, XS, nt, kc, lane);
      acc = wmma_bf16(a, b, acc);
    }
    const int n = nt * 16 + (lane & 15);
#pragma unroll
    for (int r = 0; r < 8; ++r) {
      if (G < 3) {
        dstH[(size_t)(n >> 6) * (B_ * S_ * HD_) + rowoff[r] + (n & 63)] = (bf16)acc[r];
      } else {
        outRes[rowoff[r] + n] = acc[r] + bres[n];
      }
    }
  }
}

// ---------------- Kernel 2: unary = key@Wu, softmax over S, flat [B,S,H] ----------------
__global__ __launch_bounds__(128)
void unary_softmax_kernel(const float* __restrict__ key, const float* __restrict__ Wu,
                          float* __restrict__ U) {
  const int b = blockIdx.x;
  const int h = threadIdx.x >> 5;       // wave id = head
  const int lane = threadIdx.x & 31;
  const float* kb = key + (size_t)b * S_ * E_;

  float acc[4] = {0.f, 0.f, 0.f, 0.f};
  for (int e = 0; e < E_; ++e) {
    float w = Wu[e * H_ + h];
#pragma unroll
    for (int j = 0; j < 4; ++j) {
      int s = lane + 32 * j;
      if (s < S_) acc[j] += kb[(size_t)s * E_ + e] * w;
    }
  }
  float mx = -3.0e38f;
#pragma unroll
  for (int j = 0; j < 4; ++j) { int s = lane + 32 * j; if (s < S_) mx = fmaxf(mx, acc[j]); }
  for (int off = 16; off > 0; off >>= 1) mx = fmaxf(mx, __shfl_xor(mx, off, 32));
  float sum = 0.f;
#pragma unroll
  for (int j = 0; j < 4; ++j) {
    int s = lane + 32 * j;
    if (s < S_) { acc[j] = __expf(acc[j] - mx); sum += acc[j]; }
  }
  for (int off = 16; off > 0; off >>= 1) sum += __shfl_xor(sum, off, 32);
  float inv = 1.f / sum;
#pragma unroll
  for (int j = 0; j < 4; ++j) {
    int s = lane + 32 * j;
    if (s < S_) U[((size_t)b * S_ + s) * H_ + h] = acc[j] * inv;
  }
}

// ---------------- Kernel 3: attention, one workgroup per (h,b) ----------------
__global__ __launch_bounds__(256)
void attn_kernel(const bf16* __restrict__ Qh, const bf16* __restrict__ Kh,
                 const bf16* __restrict__ Vh, const float* __restrict__ U,
                 float* __restrict__ out) {
  extern __shared__ char smem[];
  bf16* muQ    = (bf16*)smem;            // [SP][HD]  (14336 B)
  bf16* muK    = muQ + SP * HD_;         // [SP][HD]  (14336 B)
  bf16* attnP  = muQ;                    // [SP][KP]  aliases muQ+muK after GEMM1
  bf16* Vt     = (bf16*)(smem + 28672);              // [HD][KP] (16384 B)
  float* scoreF = (float*)(smem + 28672 + 16384);    // [SP][SP] f32 (50176 B)
  float* means  = (float*)(smem + 28672 + 16384 + 50176); // 128 f32

  const int i = blockIdx.x;              // i = h*B + b (head-major, matches split_heads)
  const int h = i >> 10, b = i & (B_ - 1);
  const int tid = threadIdx.x, wave = tid >> 5, lane = tid & 31;
  const size_t base = (size_t)i * S_ * HD_;

  // ---- Stage Q,K rows 0..99: async DMA global->LDS (ASYNCcnt); V^T + pads via VALU ----
#if HAVE_ASYNC_LDS
  for (int c = tid; c < 800; c += 256) {            // 800 x 16B = 100*64 bf16
    async_copy16(Qh + base + (size_t)c * 8, muQ + c * 8);
    async_copy16(Kh + base + (size_t)c * 8, muK + c * 8);
  }
#else
  for (int idx = tid; idx < S_ * HD_; idx += 256) {
    muQ[idx] = Qh[base + idx];
    muK[idx] = Kh[base + idx];
  }
#endif
  // V transposed [hd][kpos], zero-padded cols; overlaps with in-flight DMA
  for (int idx = tid; idx < HD_ * KP; idx += 256) {
    int d = idx >> 7, s = idx & 127;
    Vt[idx] = (s < S_) ? Vh[base + (size_t)s * HD_ + d] : (bf16)0.f;
  }
  // zero-pad muQ/muK rows 100..111
  for (int idx = S_ * HD_ + tid; idx < SP * HD_; idx += 256) {
    muQ[idx] = (bf16)0.f; muK[idx] = (bf16)0.f;
  }
#if HAVE_ASYNC_LDS
  wait_async0();
#endif
  __syncthreads();

  // ---- Column means over S, then subtract in place ----
  if (tid < 128) {
    const bf16* M = (tid < 64) ? muQ : muK;
    int d = tid & 63;
    float sum = 0.f;
    for (int s = 0; s < S_; ++s) sum += (float)M[s * HD_ + d];
    means[tid] = sum * (1.0f / S_);
  }
  __syncthreads();
  for (int idx = tid; idx < S_ * HD_; idx += 256) {
    int d = idx & 63;
    muQ[idx] = (bf16)((float)muQ[idx] - means[d]);
    muK[idx] = (bf16)((float)muK[idx] - means[64 + d]);
  }
  __syncthreads();

  // ---- GEMM1: wave w owns q-tile w (7 active); A-frags cached across 7 kt tiles ----
  if (wave < 7) {
    const int qt = wave;
    const bf16* arow = muQ + (size_t)(qt * 16 + (lane & 15)) * HD_;
    v16bf a0 = fragA(arow, 0, lane);
    v16bf a1 = fragA(arow, 1, lane);
    for (int kt = 0; kt < 7; ++kt) {
      v8f acc = {};
      acc = wmma_bf16(a0, fragB(muK, HD_, kt, 0, lane), acc);
      acc = wmma_bf16(a1, fragB(muK, HD_, kt, 1, lane), acc);
      float* srow = scoreF + (size_t)(qt * 16 + ((lane < 16) ? 0 : 8)) * SP
                  + kt * 16 + (lane & 15);
#pragma unroll
      for (int r = 0; r < 8; ++r) srow[(size_t)r * SP] = acc[r];
    }
  }
  __syncthreads();

  // ---- Row softmax + unary add (uw row = U_flat[i*S .. i*S+99]) -> bf16 probs ----
  const float* Urow = U + (size_t)i * S_;
  for (int q = wave; q < SP; q += 8) {
    bf16* prow = attnP + (size_t)q * KP;
    if (q < S_) {
      const float* srow = scoreF + (size_t)q * SP;
      float v[4]; float mx = -3.0e38f;
#pragma unroll
      for (int j = 0; j < 4; ++j) {
        int k = lane + 32 * j;
        v[j] = (k < S_) ? srow[k] : -3.0e38f;
        mx = fmaxf(mx, v[j]);
      }
      for (int off = 16; off > 0; off >>= 1) mx = fmaxf(mx, __shfl_xor(mx, off, 32));
      float sum = 0.f;
#pragma unroll
      for (int j = 0; j < 4; ++j) {
        int k = lane + 32 * j;
        if (k < S_) { v[j] = __expf(v[j] - mx); sum += v[j]; } else v[j] = 0.f;
      }
      for (int off = 16; off > 0; off >>= 1) sum += __shfl_xor(sum, off, 32);
      float inv = 1.f / sum;
#pragma unroll
      for (int j = 0; j < 4; ++j) {
        int k = lane + 32 * j;
        if (k < S_) prow[k] = (bf16)(v[j] * inv + Urow[k]);
      }
      if (lane < KP - S_) prow[S_ + lane] = (bf16)0.f;   // zero pad cols 100..127
    } else {
#pragma unroll
      for (int j = 0; j < 4; ++j) prow[lane + 32 * j] = (bf16)0.f;  // pad rows
    }
  }
  __syncthreads();

  // ---- GEMM2: wave w owns q-tile w; 4 A-frags cached across 4 nt tiles ----
  if (wave < 7) {
    const int qt = wave;
    const bf16* arow = attnP + (size_t)(qt * 16 + (lane & 15)) * KP;
    v16bf a0 = fragA(arow, 0, lane);
    v16bf a1 = fragA(arow, 1, lane);
    v16bf a2 = fragA(arow, 2, lane);
    v16bf a3 = fragA(arow, 3, lane);
    const size_t outbase = (size_t)b * S_ * A_ + (size_t)h * HD_;
    for (int nt = 0; nt < 4; ++nt) {
      v8f acc = {};
      acc = wmma_bf16(a0, fragB(Vt, KP, nt, 0, lane), acc);
      acc = wmma_bf16(a1, fragB(Vt, KP, nt, 1, lane), acc);
      acc = wmma_bf16(a2, fragB(Vt, KP, nt, 2, lane), acc);
      acc = wmma_bf16(a3, fragB(Vt, KP, nt, 3, lane), acc);
      const int d = nt * 16 + (lane & 15);
#pragma unroll
      for (int r = 0; r < 8; ++r) {
        int q = qt * 16 + r + ((lane < 16) ? 0 : 8);
        if (q < S_) {
          float* dst = out + outbase + (size_t)q * A_ + d;
          *dst += acc[r];               // residual already written by proj_kernel<3>
        }
      }
    }
  }
}

// ---------------- Launch ----------------
extern "C" void kernel_launch(void* const* d_in, const int* in_sizes, int n_in,
                              void* d_out, int out_size, void* d_ws, size_t ws_size,
                              hipStream_t stream) {
  const float* query = (const float*)d_in[0];
  const float* key   = (const float*)d_in[1];
  const float* value = (const float*)d_in[2];
  const float* Wq    = (const float*)d_in[3];
  const float* Wk    = (const float*)d_in[4];
  const float* Wv    = (const float*)d_in[5];
  const float* Wu    = (const float*)d_in[6];
  const float* Wres  = (const float*)d_in[7];
  const float* bres  = (const float*)d_in[8];
  float* out = (float*)d_out;

  char* ws = (char*)d_ws;
  const size_t qkvElems = (size_t)H_ * B_ * S_ * HD_;   // 26,214,400
  bf16* Qh = (bf16*)ws;
  bf16* Kh = Qh + qkvElems;
  bf16* Vh = Kh + qkvElems;
  float* U = (float*)(ws + 3 * qkvElems * sizeof(bf16));

  const size_t projLds = (size_t)(128 + 256) * XS * sizeof(bf16);   // 104448 B
  const size_t attnLds = 28672 + 16384 + 50176 + 512;               //  95744 B

  proj_kernel<0><<<800, 256, projLds, stream>>>(query, Wq,   bres, Qh, out);
  proj_kernel<1><<<800, 256, projLds, stream>>>(key,   Wk,   bres, Kh, out);
  proj_kernel<2><<<800, 256, projLds, stream>>>(value, Wv,   bres, Vh, out);
  proj_kernel<3><<<800, 256, projLds, stream>>>(query, Wres, bres, (bf16*)nullptr, out);
  unary_softmax_kernel<<<B_, 128, 0, stream>>>(key, Wu, U);
  attn_kernel<<<H_ * B_, 256, attnLds, stream>>>(Qh, Kh, Vh, U, out);
}
